// SAKELayer_20564303413684
// MI455X (gfx1250) — matrix operand order repaired and checked by reading, not compiled, not run on hardware
//
#include <hip/hip_runtime.h>
#include <hip/hip_bf16.h>
#include <math.h>

// ---------------- types ----------------
typedef __attribute__((ext_vector_type(16))) _Float16 v16h;
typedef __attribute__((ext_vector_type(8)))  float    v8f;

#define Bsz 16
#define Nn  64
#define Ff  64
#define Hh  4
#define Rr  50
#define Cc  256

__device__ __forceinline__ float siluf(float v) { return v / (1.0f + __expf(-v)); }
__device__ __forceinline__ float tanh_fast(float x) {
  float e = __expf(2.0f * x);
  return 1.0f - 2.0f / (e + 1.0f);
}

// ---- WMMA fragment loaders (CDNA5 wave32, 16-bit, 16x16x32) ----
// A 16x32 (MxK): lane L -> row M=L&15, half h=L>>4.
//   elements 0..15 = K = h*16 + {0..15}   (contiguous)
__device__ __forceinline__ v16h load_a(const _Float16* A, int lda, int row0, int k0, int lane) {
  const int m = lane & 15, hf = lane >> 4;
  v16h a;
#pragma unroll
  for (int e = 0; e < 16; ++e) {
    const int vg = e >> 1, pos = e & 1;
    const int K = hf * 16 + vg * 2 + pos;
    a[e] = A[(size_t)(row0 + m) * lda + k0 + K];
  }
  return a;
}
// B fragments are pre-swizzled by k_cvt: tile (kt,nt), lane l, 16 halves contiguous.
// Logical value at element e: W[kt*32 + (l>>4)*16 + e][nt*16 + (l&15)]
__device__ __forceinline__ v16h load_b_sw(const _Float16* Wsw, int NT, int kt, int nt, int lane) {
  const _Float16* p = Wsw + ((size_t)(kt * NT + nt) * 32 + lane) * 16;
  v16h b;
#pragma unroll
  for (int e = 0; e < 16; ++e) b[e] = p[e];   // 32B contiguous -> 2x global_load_b128
  return b;
}
__device__ __forceinline__ v8f wmma_f16(v16h a, v16h b, v8f c) {
  return __builtin_amdgcn_wmma_f32_16x16x32_f16(false, a, false, b, (short)0, c, false, false);
}
// C/D 16x16 f32: row = r + 8*(lane>>4), col = lane&15.

// ---------------- K0: weight convert/pad to f16, pre-swizzled B-fragment order ----
__global__ void k_cvt(const float* __restrict__ W_in, const float* __restrict__ W_e1,
                      const float* __restrict__ W_e2, const float* __restrict__ W_xmix,
                      _Float16* Win_h, _Float16* We1_h, _Float16* We2_h, _Float16* Wxm_h) {
  const int idx = blockIdx.x * blockDim.x + threadIdx.x;
  const int stride = gridDim.x * blockDim.x;
  // W_in: K=128 (KT=4), N=64 pad from 50 (NT=4)
  for (int t = idx; t < 128 * 64; t += stride) {
    const int e = t & 15, l = (t >> 4) & 31, tile = t >> 9;
    const int nt = tile & 3, kt = tile >> 2;
    const int K = kt * 32 + (l >> 4) * 16 + e;
    const int n = nt * 16 + (l & 15);
    Win_h[t] = (_Float16)(n < Rr ? W_in[K * Rr + n] : 0.0f);
  }
  // W_e1: K pad 179->192 (KT=6), N=64 (NT=4)
  for (int t = idx; t < 192 * 64; t += stride) {
    const int e = t & 15, l = (t >> 4) & 31, tile = t >> 9;
    const int nt = tile & 3, kt = tile >> 2;
    const int K = kt * 32 + (l >> 4) * 16 + e;
    const int n = nt * 16 + (l & 15);
    We1_h[t] = (_Float16)(K < 179 ? W_e1[K * 64 + n] : 0.0f);
  }
  // W_e2: K=64 (KT=2), N=64 (NT=4)
  for (int t = idx; t < 64 * 64; t += stride) {
    const int e = t & 15, l = (t >> 4) & 31, tile = t >> 9;
    const int nt = tile & 3, kt = tile >> 2;
    const int K = kt * 32 + (l >> 4) * 16 + e;
    const int n = nt * 16 + (l & 15);
    We2_h[t] = (_Float16)W_e2[K * 64 + n];
  }
  // W_xmix: K=256 (KT=8), N=256 (NT=16)
  for (int t = idx; t < 256 * 256; t += stride) {
    const int e = t & 15, l = (t >> 4) & 31, tile = t >> 9;
    const int nt = tile & 15, kt = tile >> 4;
    const int K = kt * 32 + (l >> 4) * 16 + e;
    const int n = nt * 16 + (l & 15);
    Wxm_h[t] = (_Float16)W_xmix[K * 256 + n];
  }
}

// ---------------- K1: edge model (1 wave per 16-edge tile) ----------------
__global__ __launch_bounds__(32) void k_edge(
    const float* __restrict__ h, const float* __restrict__ x,
    const _Float16* __restrict__ Win_h, const float* __restrict__ b_in,
    const float* __restrict__ rbf_means, const float* __restrict__ rbf_betas,
    const _Float16* __restrict__ We1_h, const float* __restrict__ b_e1,
    const _Float16* __restrict__ We2_h, const float* __restrict__ b_e2,
    const float* __restrict__ W_sem, const float* __restrict__ b_sem,
    float* __restrict__ he_g, float* __restrict__ d_g, float* __restrict__ sem_g) {
  const int lane = threadIdx.x & 31;
  const int bid = blockIdx.x;
  const int jt = bid & 3, i = (bid >> 2) & 63, b = bid >> 8;
  const int j0 = jt * 16;
  const int bi = b * Nn + i;

  __shared__ _Float16 A2[16][192];   // [h_j | h_i | rbf*h_mid | d | pad]
  __shared__ _Float16 A3[16][64];
  __shared__ float he_s[16][64];
  __shared__ float d_s[16];

  // prefetch f16 weight panels into cache (global_prefetch_b8)
  __builtin_prefetch(Win_h + lane * 256, 0, 1);
  __builtin_prefetch(We1_h + lane * 384, 0, 1);
  __builtin_prefetch(We2_h + lane * 128, 0, 1);

  if (lane < 16) {
    const int j = j0 + lane;
    float dx = x[(b * Nn + j) * 3 + 0] - x[(b * Nn + i) * 3 + 0];
    float dy = x[(b * Nn + j) * 3 + 1] - x[(b * Nn + i) * 3 + 1];
    float dz = x[(b * Nn + j) * 3 + 2] - x[(b * Nn + i) * 3 + 2];
    float r2 = dx * dx + dy * dy + dz * dz;
    d_s[lane] = sqrtf(fmaxf(r2, 0.0f) + 1e-5f);
  }
  for (int t = lane; t < 16 * 192; t += 32) {
    const int m = t / 192, k = t - m * 192;
    float val;
    if (k < 64)       val = h[(b * Nn + (j0 + m)) * Ff + k];
    else if (k < 128) val = h[(b * Nn + i) * Ff + (k - 64)];
    else              val = 0.0f;
    A2[m][k] = (_Float16)val;
  }
  __syncthreads();

  // h_mid = h_cat @ W_in + b_in, then rbf smearing -> A2 cols 128..177
  {
    v16h afr[4];
#pragma unroll
    for (int kc = 0; kc < 4; ++kc) afr[kc] = load_a(&A2[0][0], 192, 0, kc * 32, lane);
    for (int nt = 0; nt < 4; ++nt) {
      v8f acc = {};
#pragma unroll
      for (int kc = 0; kc < 4; ++kc)
        acc = wmma_f16(afr[kc], load_b_sw(Win_h, 4, kc, nt, lane), acc);
      const int col = nt * 16 + (lane & 15);
#pragma unroll
      for (int r = 0; r < 8; ++r) {
        const int m = r + 8 * (lane >> 4);
        if (col < Rr) {
          float hm = acc[r] + b_in[col];
          float d = d_s[m];
          float cut = (d < 5.0f) ? 0.5f * (__cosf(d * 0.62831853071795865f) + 1.0f) : 0.0f;
          float tt = __expf(-d) - rbf_means[col];
          float rbf = cut * __expf(-rbf_betas[col] * tt * tt);
          A2[m][128 + col] = (_Float16)(rbf * hm);  // disjoint from K=0..127 reads above
        }
      }
    }
  }
  if (lane < 16) A2[lane][178] = (_Float16)d_s[lane];  // d column
  __syncthreads();

  // h_e = silu([h_cat, rbf*h_mid, d] @ W_e1 + b_e1) @ W_e2 + b_e2
  {
    v16h afr[6];
#pragma unroll
    for (int kc = 0; kc < 6; ++kc) afr[kc] = load_a(&A2[0][0], 192, 0, kc * 32, lane);
    for (int nt = 0; nt < 4; ++nt) {
      v8f acc = {};
#pragma unroll
      for (int kc = 0; kc < 6; ++kc)
        acc = wmma_f16(afr[kc], load_b_sw(We1_h, 4, kc, nt, lane), acc);
      const int col = nt * 16 + (lane & 15);
#pragma unroll
      for (int r = 0; r < 8; ++r) {
        const int m = r + 8 * (lane >> 4);
        A3[m][col] = (_Float16)siluf(acc[r] + b_e1[col]);
      }
    }
  }
  __syncthreads();
  {
    v16h afr[2];
#pragma unroll
    for (int kc = 0; kc < 2; ++kc) afr[kc] = load_a(&A3[0][0], 64, 0, kc * 32, lane);
    for (int nt = 0; nt < 4; ++nt) {
      v8f acc = {};
#pragma unroll
      for (int kc = 0; kc < 2; ++kc)
        acc = wmma_f16(afr[kc], load_b_sw(We2_h, 4, kc, nt, lane), acc);
      const int col = nt * 16 + (lane & 15);
#pragma unroll
      for (int r = 0; r < 8; ++r) {
        const int m = r + 8 * (lane >> 4);
        float val = acc[r] + b_e2[col];
        he_s[m][col] = val;
        he_g[((size_t)bi * Nn + (j0 + m)) * Ff + col] = val;
      }
    }
  }
  __syncthreads();

  // sem logits: celu(h_e @ W_sem + b_sem, alpha=2)
  {
    const int m = lane & 15;
    const int hbase = lane >> 4;  // heads hbase and hbase+2
    const int j = j0 + m;
#pragma unroll
    for (int hh = 0; hh < 2; ++hh) {
      const int hd = hbase + 2 * hh;
      float s = b_sem[hd];
      for (int f = 0; f < Ff; ++f) s += he_s[m][f] * W_sem[f * Hh + hd];
      float cel = (s > 0.0f) ? s : 2.0f * (__expf(0.5f * s) - 1.0f);
      sem_g[((size_t)bi * Nn + j) * Hh + hd] = cel;
    }
  }
  if (lane < 16) d_g[(size_t)bi * Nn + j0 + lane] = d_s[lane];
}

// ---------------- K2: dual softmax over neighbors ----------------
__device__ __forceinline__ float red_max(float v, float* red) {
  const int t = threadIdx.x;
  red[t] = v; __syncthreads();
  for (int s = 32; s > 0; s >>= 1) { if (t < s) red[t] = fmaxf(red[t], red[t + s]); __syncthreads(); }
  float r = red[0]; __syncthreads();
  return r;
}
__device__ __forceinline__ float red_sum(float v, float* red) {
  const int t = threadIdx.x;
  red[t] = v; __syncthreads();
  for (int s = 32; s > 0; s >>= 1) { if (t < s) red[t] += red[t + s]; __syncthreads(); }
  float r = red[0]; __syncthreads();
  return r;
}
__global__ __launch_bounds__(64) void k_att(const float* __restrict__ sem_g,
                                            const float* __restrict__ d_g,
                                            const float* __restrict__ log_gamma,
                                            float* __restrict__ att_g) {
  const int bi = blockIdx.x;
  const int i = bi & 63;
  const int j = threadIdx.x;
  __shared__ float red[64];
  const float d = d_g[(size_t)bi * Nn + j];
  const float mask = (j == i) ? 1e5f : 0.0f;
  float att[Hh];
#pragma unroll
  for (int hd = 0; hd < Hh; ++hd) {
    float ls = sem_g[((size_t)bi * Nn + j) * Hh + hd] - mask;
    float le = -(d + mask) * __expf(log_gamma[hd]);
    float m1 = red_max(ls, red);
    float es = __expf(ls - m1);
    float s1 = red_sum(es, red);
    float m2 = red_max(le, red);
    float ee = __expf(le - m2);
    float s2 = red_sum(ee, red);
    float a = (es / s1) * (ee / s2);
    float s3 = red_sum(a, red);
    att[hd] = a / s3;
  }
#pragma unroll
  for (int hd = 0; hd < Hh; ++hd)
    att_g[((size_t)bi * Nn + j) * Hh + hd] = att[hd];
}

// ---------------- K3: h_e_att, coeff = tanh(h_e_att @ W_xmix), spatial aggregates ----
__global__ __launch_bounds__(128) void k_spatial(
    const float* __restrict__ x, const float* __restrict__ he_g,
    const float* __restrict__ att_g, const float* __restrict__ d_g,
    const _Float16* __restrict__ Wxm_h, const float* __restrict__ W_vmix,
    float* __restrict__ hagg_g, float* __restrict__ hcp_g, float* __restrict__ dv_g) {
  const int bi = blockIdx.x;
  const int b = bi >> 6, i = bi & 63;
  const int tid = threadIdx.x, lane = tid & 31, wv = tid >> 5;

  __shared__ _Float16 he_att[64][256];  // rows = neighbor j, cols = c (f*4+h)
  __shared__ float xhat[64][3];
  __shared__ float comb[256][3];
  __shared__ float dv_s[3];

  // prefetch the 128KB W_xmix f16 panel (global_prefetch_b8, L2-resident)
  for (int r = tid; r < 256; r += 128) __builtin_prefetch(Wxm_h + (size_t)r * 256, 0, 1);

  if (tid < 64) {
    const int j = tid;
    const float d = d_g[(size_t)bi * Nn + j];
    const float inv = 1.0f / (d + 1e-5f);
#pragma unroll
    for (int k = 0; k < 3; ++k)
      xhat[j][k] = (x[(b * Nn + j) * 3 + k] - x[(b * Nn + i) * 3 + k]) * inv;
  }
  for (int t = tid; t < 256 * 3; t += 128) (&comb[0][0])[t] = 0.0f;
  if (tid < 3) dv_s[tid] = 0.0f;
  for (int t = tid; t < 64 * 256; t += 128) {
    const int j = t >> 8, c = t & 255;
    const int f = c >> 2, hd = c & 3;
    (&he_att[0][0])[t] =
        (_Float16)(he_g[((size_t)bi * Nn + j) * Ff + f] * att_g[((size_t)bi * Nn + j) * Hh + hd]);
  }
  __syncthreads();

  // h_agg = sum_j h_e_att
  for (int c = tid; c < 256; c += 128) {
    float s = 0.0f;
    for (int j = 0; j < 64; ++j) s += (float)he_att[j][c];
    hagg_g[(size_t)bi * Cc + c] = s;
  }

  // coeff = tanh(h_e_att @ W_xmix); fold into comb via LDS atomics
  const int mrow0 = wv * 16;
  {
    v16h afr[8];  // hoisted A fragments: reused across all 16 N-tiles
#pragma unroll
    for (int kc = 0; kc < 8; ++kc)
      afr[kc] = load_a(&he_att[0][0], 256, mrow0, kc * 32, lane);
    for (int nt = 0; nt < 16; ++nt) {
      v8f acc = {};
#pragma unroll
      for (int kc = 0; kc < 8; ++kc)
        acc = wmma_f16(afr[kc], load_b_sw(Wxm_h, 16, kc, nt, lane), acc);
      const int col = nt * 16 + (lane & 15);
#pragma unroll
      for (int r = 0; r < 8; ++r) {
        const int j = mrow0 + r + 8 * (lane >> 4);
        const float cf = tanh_fast(acc[r]);
#pragma unroll
        for (int k = 0; k < 3; ++k)
          atomicAdd(&comb[col][k], cf * xhat[j][k]);
      }
    }
  }
  __syncthreads();

  for (int c = tid; c < 256; c += 128) {
    float cs[3], hc = 0.0f;
#pragma unroll
    for (int k = 0; k < 3; ++k) { cs[k] = comb[c][k] * (1.0f / 64.0f); hc += cs[k] * cs[k]; }
    hcp_g[(size_t)bi * Cc + c] = hc;
    const float wmix = W_vmix[c];
#pragma unroll
    for (int k = 0; k < 3; ++k) atomicAdd(&dv_s[k], cs[k] * wmix);
  }
  __syncthreads();
  if (tid < 3) dv_g[(size_t)bi * 3 + tid] = dv_s[tid];
}

// ---------------- K4: node update + velocity/coordinate update ----------------
__global__ __launch_bounds__(64) void k_node(
    const float* __restrict__ h, const float* __restrict__ x, const float* __restrict__ v,
    const float* __restrict__ hagg_g, const float* __restrict__ hcp_g, const float* __restrict__ dv_g,
    const float* __restrict__ W_post1, const float* __restrict__ b_post1,
    const float* __restrict__ W_post2, const float* __restrict__ b_post2,
    const float* __restrict__ W_node1, const float* __restrict__ b_node1,
    const float* __restrict__ W_node2, const float* __restrict__ b_node2,
    const float* __restrict__ W_vel1, const float* __restrict__ b_vel1,
    const float* __restrict__ W_vel2, float* __restrict__ out) {
  const int bi = blockIdx.x;
  const int f = threadIdx.x;
  __shared__ float s_in[384];
  __shared__ float s_a[64];
  __shared__ float s_b[64];
  __shared__ float gsh;

  for (int t = f; t < 256; t += 64) s_in[t] = hcp_g[(size_t)bi * Cc + t];
  __syncthreads();
  float acc = b_post1[f];
  for (int k = 0; k < 256; ++k) acc += s_in[k] * W_post1[k * 64 + f];
  s_a[f] = siluf(acc);
  __syncthreads();
  acc = b_post2[f];
  for (int k = 0; k < 64; ++k) acc += s_a[k] * W_post2[k * 64 + f];
  const float hcomb = siluf(acc);
  __syncthreads();

  s_in[f] = h[(size_t)bi * Ff + f];
  for (int t = f; t < 256; t += 64) s_in[64 + t] = hagg_g[(size_t)bi * Cc + t];
  s_in[320 + f] = hcomb;
  __syncthreads();
  acc = b_node1[f];
  for (int k = 0; k < 384; ++k) acc += s_in[k] * W_node1[k * 64 + f];
  s_a[f] = siluf(acc);
  __syncthreads();
  acc = b_node2[f];
  for (int k = 0; k < 64; ++k) acc += s_a[k] * W_node2[k * 64 + f];
  const float hout = s_in[f] + siluf(acc);
  out[(size_t)bi * Ff + f] = hout;
  s_b[f] = hout;
  __syncthreads();
  acc = b_vel1[f];
  for (int k = 0; k < 64; ++k) acc += s_b[k] * W_vel1[k * 64 + f];
  s_a[f] = siluf(acc);
  __syncthreads();
  if (f == 0) {
    float z = 0.0f;
    for (int k = 0; k < 64; ++k) z += s_a[k] * W_vel2[k];
    gsh = 2.0f / (1.0f + __expf(-z));
  }
  __syncthreads();
  if (f < 3) {
    const float dv = dv_g[(size_t)bi * 3 + f];
    const float vout = dv + gsh * v[(size_t)bi * 3 + f];
    const float xout = x[(size_t)bi * 3 + f] + vout;
    out[(size_t)Bsz * Nn * Ff + (size_t)bi * 3 + f] = xout;                        // x_out
    out[(size_t)Bsz * Nn * Ff + (size_t)Bsz * Nn * 3 + (size_t)bi * 3 + f] = vout; // v_out
  }
}

// ---------------- host launch ----------------
extern "C" void kernel_launch(void* const* d_in, const int* in_sizes, int n_in,
                              void* d_out, int out_size, void* d_ws, size_t ws_size,
                              hipStream_t stream) {
  (void)in_sizes; (void)n_in; (void)out_size; (void)ws_size;
  const float* h        = (const float*)d_in[0];
  const float* x        = (const float*)d_in[1];
  const float* v        = (const float*)d_in[2];
  const float* W_in     = (const float*)d_in[3];
  const float* b_in     = (const float*)d_in[4];
  const float* rbf_mean = (const float*)d_in[5];
  const float* rbf_beta = (const float*)d_in[6];
  const float* W_e1     = (const float*)d_in[7];
  const float* b_e1     = (const float*)d_in[8];
  const float* W_e2     = (const float*)d_in[9];
  const float* b_e2     = (const float*)d_in[10];
  const float* W_sem    = (const float*)d_in[11];
  const float* b_sem    = (const float*)d_in[12];
  const float* log_gam  = (const float*)d_in[13];
  const float* W_xmix   = (const float*)d_in[14];
  const float* W_post1  = (const float*)d_in[15];
  const float* b_post1  = (const float*)d_in[16];
  const float* W_post2  = (const float*)d_in[17];
  const float* b_post2  = (const float*)d_in[18];
  const float* W_node1  = (const float*)d_in[19];
  const float* b_node1  = (const float*)d_in[20];
  const float* W_node2  = (const float*)d_in[21];
  const float* b_node2  = (const float*)d_in[22];
  const float* W_vel1   = (const float*)d_in[23];
  const float* b_vel1   = (const float*)d_in[24];
  const float* W_vel2   = (const float*)d_in[25];
  const float* W_vmix   = (const float*)d_in[26];
  float* out = (float*)d_out;

  char* ws = (char*)d_ws;
  size_t off = 0;
  auto take = [&](size_t bytes) -> void* {
    void* p = ws + off;
    off = (off + bytes + 255) & ~(size_t)255;
    return p;
  };
  _Float16* Win_h = (_Float16*)take(128 * 64 * 2);
  _Float16* We1_h = (_Float16*)take(192 * 64 * 2);
  _Float16* We2_h = (_Float16*)take(64 * 64 * 2);
  _Float16* Wxm_h = (_Float16*)take(256 * 256 * 2);
  float* he_g  = (float*)take((size_t)Bsz * Nn * Nn * Ff * 4);
  float* d_g   = (float*)take((size_t)Bsz * Nn * Nn * 4);
  float* sem_g = (float*)take((size_t)Bsz * Nn * Nn * Hh * 4);
  float* att_g = (float*)take((size_t)Bsz * Nn * Nn * Hh * 4);
  float* hagg  = (float*)take((size_t)Bsz * Nn * Cc * 4);
  float* hcp   = (float*)take((size_t)Bsz * Nn * Cc * 4);
  float* dv    = (float*)take((size_t)Bsz * Nn * 3 * 4);

  k_cvt<<<64, 256, 0, stream>>>(W_in, W_e1, W_e2, W_xmix, Win_h, We1_h, We2_h, Wxm_h);
  k_edge<<<Bsz * Nn * 4, 32, 0, stream>>>(h, x, Win_h, b_in, rbf_mean, rbf_beta,
                                          We1_h, b_e1, We2_h, b_e2, W_sem, b_sem,
                                          he_g, d_g, sem_g);
  k_att<<<Bsz * Nn, 64, 0, stream>>>(sem_g, d_g, log_gam, att_g);
  k_spatial<<<Bsz * Nn, 128, 0, stream>>>(x, he_g, att_g, d_g, Wxm_h, W_vmix,
                                          hagg, hcp, dv);
  k_node<<<Bsz * Nn, 64, 0, stream>>>(h, x, v, hagg, hcp, dv,
                                      W_post1, b_post1, W_post2, b_post2,
                                      W_node1, b_node1, W_node2, b_node2,
                                      W_vel1, b_vel1, W_vel2, out);
}
